// GATNet_72868415144432
// MI455X (gfx1250) — compile-verified, matrix operand order checked by reading, not run on hardware
//
#include <hip/hip_runtime.h>
#include <hip/hip_bf16.h>
#include <stdint.h>

// ---------------- problem constants (match reference) ----------------
#define NN      50000
#define EE      800000
#define FIN     78
#define H1      10
#define HF1     780      // H1*FIN
#define OUT2    128
#define GG      512
#define ET      (EE + NN)   // edges + self loops
#define KP1     96          // FIN padded to mult of 32
#define KP2     800         // HF1 padded to mult of 32
#define KPF     128         // OUT2 (already mult of 32)
#define NEG_SLOPE 0.2f
#define ENC_NEG_INF ((int)0x807FFFFF)  // ordered-int encoding of -inf

typedef __bf16  v16bf __attribute__((ext_vector_type(16)));
typedef __bf16  v8bf  __attribute__((ext_vector_type(8)));
typedef float   v8f   __attribute__((ext_vector_type(8)));
typedef int     v4i   __attribute__((ext_vector_type(4)));

#if defined(__has_builtin)
#  if __has_builtin(__builtin_amdgcn_global_load_async_to_lds_b128)
#    define HAVE_ASYNC_LDS 1
#  endif
#endif
#ifndef HAVE_ASYNC_LDS
#  define HAVE_ASYNC_LDS 0
#endif

// ---------------- helpers ----------------
__device__ __forceinline__ unsigned short f2bf(float f) {
  unsigned u = __float_as_uint(f);
  u += 0x7FFFu + ((u >> 16) & 1u);          // round to nearest even
  return (unsigned short)(u >> 16);
}
__device__ __forceinline__ int enc_f32(float f) {    // order-preserving int
  int i = __float_as_int(f);
  return (i >= 0) ? i : (i ^ 0x7FFFFFFF);
}
__device__ __forceinline__ float dec_f32(int i) {
  unsigned b = (i >= 0) ? (unsigned)i : ((unsigned)i ^ 0x7FFFFFFFu);
  return __uint_as_float(b);
}

// ---------------- generic fills ----------------
__global__ void k_fill_i32(int* p, int v, size_t n) {
  size_t i = (size_t)blockIdx.x * blockDim.x + threadIdx.x;
  if (i < n) p[i] = v;
}
__global__ void k_fill_f32(float* p, float v, size_t n) {
  size_t i = (size_t)blockIdx.x * blockDim.x + threadIdx.x;
  if (i < n) p[i] = v;
}

// ---------------- f32 -> bf16 conversions ----------------
// A: [M,K] row-major f32 -> [M,Kpad] bf16, zero-padded K
__global__ void k_conv_a(const float* __restrict__ in, unsigned short* __restrict__ out,
                         int M, int K, int Kpad) {
  size_t idx = (size_t)blockIdx.x * blockDim.x + threadIdx.x;
  size_t total = (size_t)M * Kpad;
  if (idx >= total) return;
  int k = (int)(idx % Kpad);
  size_t m = idx / Kpad;
  out[idx] = (k < K) ? f2bf(in[m * K + k]) : (unsigned short)0;
}
// W: [K,Nc] row-major f32 -> Bt: [Nc,Kpad] bf16 (transposed, zero-padded K)
__global__ void k_conv_bt(const float* __restrict__ W, unsigned short* __restrict__ out,
                          int K, int Nc, int Kpad) {
  size_t idx = (size_t)blockIdx.x * blockDim.x + threadIdx.x;
  size_t total = (size_t)Nc * Kpad;
  if (idx >= total) return;
  int k = (int)(idx % Kpad);
  size_t n = idx / Kpad;
  out[idx] = (k < K) ? f2bf(W[(size_t)k * Nc + n]) : (unsigned short)0;
}

// ---------------- WMMA bf16 GEMM: C[M,Nc] = A[M,Kpad] * Bt[Nc,Kpad]^T ----------------
// Block = 8 waves. Each block: stage a 64-column Bt panel in LDS (async copy),
// each wave computes a 16x64 C tile (4 accumulators), A fragment reused 4x,
// B fragments served from LDS (ds_load_b128), V_WMMA_F32_16X16X32_BF16.
__device__ __forceinline__ v16bf load_frag(const unsigned short* p) {
  // per ISA 7.12.2 (16-bit 16x32 fragment): this lane holds K = [base, base+8)
  // and [base+16, base+24) as two contiguous 16B chunks.
  v8bf lo = *reinterpret_cast<const v8bf*>(p);
  v8bf hi = *reinterpret_cast<const v8bf*>(p + 16);
  v16bf r;
#pragma unroll
  for (int i = 0; i < 8; ++i) { r[i] = lo[i]; r[8 + i] = hi[i]; }
  return r;
}

__global__ void k_wmma_gemm_bf16(const unsigned short* __restrict__ A,
                                 const unsigned short* __restrict__ Bt,
                                 float* __restrict__ C,
                                 const float* __restrict__ bias,
                                 int M, int Nc, int Kpad, int applyRelu) {
  extern __shared__ unsigned short smem[];   // [64][Kpad] bf16 Bt panel
  const int tid  = threadIdx.x;
  const int wave = tid >> 5;
  const int lane = tid & 31;
  const int half = lane >> 4;
  const int l16  = lane & 15;
  const int nBase = blockIdx.y * 64;

  // ---- stage Bt panel into LDS: 16B per thread per iteration ----
  const int chunksTotal = (64 * Kpad) / 8;         // Kpad is a multiple of 8
  for (int c = tid; c < chunksTotal; c += blockDim.x) {
    const int elem   = c * 8;
    const int nLocal = elem / Kpad;
    const int k      = elem % Kpad;
    int grow = nBase + nLocal; if (grow > Nc - 1) grow = Nc - 1;  // dup, masked at store
    const unsigned short* gsrc = Bt + (size_t)grow * Kpad + k;
    unsigned short*       ldst = &smem[nLocal * Kpad + k];
#if HAVE_ASYNC_LDS
    typedef __attribute__((address_space(1))) v4i g_v4i;   // global int4*
    typedef __attribute__((address_space(3))) v4i l_v4i;   // LDS int4*
    __builtin_amdgcn_global_load_async_to_lds_b128(
        (g_v4i*)(size_t)gsrc,
        (l_v4i*)(size_t)(unsigned)(size_t)ldst,   // low 32 bits = LDS offset
        0, 0);
#else
    *reinterpret_cast<v8bf*>(ldst) = *reinterpret_cast<const v8bf*>(gsrc);
#endif
  }
#if HAVE_ASYNC_LDS
#  if __has_builtin(__builtin_amdgcn_s_wait_asynccnt)
  __builtin_amdgcn_s_wait_asynccnt(0);
#  else
  asm volatile("s_wait_asynccnt 0" ::: "memory");
#  endif
#endif
  __syncthreads();

  // ---- compute this wave's 16x64 tile ----
  const int m0 = blockIdx.x * 128 + wave * 16;
  int arow = m0 + l16; if (arow > M - 1) arow = M - 1;   // clamp, mask at store
  const unsigned short* ap = A + (size_t)arow * Kpad + half * 8;

  v8f acc0 = {}, acc1 = {}, acc2 = {}, acc3 = {};
  const unsigned short* bp = &smem[l16 * Kpad + half * 8];
  const int bstep = 16 * Kpad;

  for (int k = 0; k < Kpad; k += 32) {
    v16bf a  = load_frag(ap + k);
    v16bf b0 = load_frag(bp + k);
    v16bf b1 = load_frag(bp + bstep + k);
    v16bf b2 = load_frag(bp + 2 * bstep + k);
    v16bf b3 = load_frag(bp + 3 * bstep + k);
    acc0 = __builtin_amdgcn_wmma_f32_16x16x32_bf16(false, a, false, b0, (short)0, acc0, false, false);
    acc1 = __builtin_amdgcn_wmma_f32_16x16x32_bf16(false, a, false, b1, (short)0, acc1, false, false);
    acc2 = __builtin_amdgcn_wmma_f32_16x16x32_bf16(false, a, false, b2, (short)0, acc2, false, false);
    acc3 = __builtin_amdgcn_wmma_f32_16x16x32_bf16(false, a, false, b3, (short)0, acc3, false, false);
  }

  // ---- store (D layout: VGPR r -> M = r + 8*half, N = l16) ----
  v8f accs[4] = {acc0, acc1, acc2, acc3};
#pragma unroll
  for (int s = 0; s < 4; ++s) {
    const int n = nBase + s * 16 + l16;
    if (n < Nc) {
      const float bv = bias ? bias[n] : 0.0f;
#pragma unroll
      for (int r = 0; r < 8; ++r) {
        const int m = m0 + r + half * 8;
        if (m < M) {
          float v = accs[s][r] + bv;
          if (applyRelu) v = fmaxf(v, 0.0f);
          C[(size_t)m * Nc + n] = v;
        }
      }
    }
  }
}

// ---------------- attention projections: alpha[n,h] = sum_f h[n,h,f]*a[h,f] ----------------
__global__ void k_alpha_proj(const float* __restrict__ h, const float* __restrict__ a,
                             float* __restrict__ out, int N, int H, int F) {
  size_t idx = (size_t)blockIdx.x * blockDim.x + threadIdx.x;
  size_t total = (size_t)N * H;
  if (idx >= total) return;
  int hh = (int)(idx % H);
  size_t n = idx / H;
  const float* hp = h + n * (size_t)(H * F) + (size_t)hh * F;
  const float* apv = a + (size_t)hh * F;
  float s = 0.0f;
  for (int f = 0; f < F; ++f) s += hp[f] * apv[f];
  out[idx] = s;
}

// ---------------- edge pass 1: leaky-relu logits + segment max ----------------
__global__ void k_edge_logits(const int* __restrict__ ei,
                              const float* __restrict__ asrc, const float* __restrict__ adst,
                              float* __restrict__ ebuf, int* __restrict__ maxb, int H) {
  size_t idx = (size_t)blockIdx.x * blockDim.x + threadIdx.x;
  size_t total = (size_t)ET * H;
  if (idx >= total) return;
  int h = (int)(idx % H);
  size_t e = idx / H;
  int s, d;
  if (e < (size_t)EE) { s = ei[e]; d = ei[(size_t)EE + e]; }
  else                { s = d = (int)(e - EE); }
  float v = asrc[(size_t)s * H + h] + adst[(size_t)d * H + h];
  v = (v > 0.0f) ? v : NEG_SLOPE * v;
  ebuf[idx] = v;
  atomicMax(&maxb[(size_t)d * H + h], enc_f32(v));
}

// ---------------- edge pass 2: exp(e - max) + segment sum ----------------
__global__ void k_edge_exp(const int* __restrict__ ei,
                           float* __restrict__ ebuf, const int* __restrict__ maxb,
                           float* __restrict__ denom, int H) {
  size_t idx = (size_t)blockIdx.x * blockDim.x + threadIdx.x;
  size_t total = (size_t)ET * H;
  if (idx >= total) return;
  int h = (int)(idx % H);
  size_t e = idx / H;
  int d = (e < (size_t)EE) ? ei[(size_t)EE + e] : (int)(e - EE);
  float m = dec_f32(maxb[(size_t)d * H + h]);
  if (!__builtin_isfinite(m)) m = 0.0f;
  float ex = __expf(ebuf[idx] - m);
  ebuf[idx] = ex;
  atomicAdd(&denom[(size_t)d * H + h], ex);
}

// ---------------- edge pass 3: alpha-weighted scatter (grid-stride) ----------------
__global__ void k_edge_scatter(const int* __restrict__ ei,
                               const float* __restrict__ ebuf, const float* __restrict__ denom,
                               const float* __restrict__ hsrc, float* __restrict__ agg,
                               int H, int F) {
  const size_t HF = (size_t)H * F;
  const size_t total = (size_t)ET * HF;
  const size_t stride = (size_t)gridDim.x * blockDim.x;
  for (size_t idx = (size_t)blockIdx.x * blockDim.x + threadIdx.x; idx < total; idx += stride) {
    size_t e = idx / HF;
    int hf = (int)(idx % HF);
    int h  = hf / F;
    int s, d;
    if (e < (size_t)EE) { s = ei[e]; d = ei[(size_t)EE + e]; }
    else                { s = d = (int)(e - EE); }
    float alpha = ebuf[e * H + h] / (denom[(size_t)d * H + h] + 1e-16f);
    atomicAdd(&agg[(size_t)d * HF + hf], hsrc[(size_t)s * HF + hf] * alpha);
  }
}

// ---------------- bias + relu (in place) ----------------
__global__ void k_bias_relu(float* __restrict__ buf, const float* __restrict__ bias,
                            int N, int HF) {
  size_t idx = (size_t)blockIdx.x * blockDim.x + threadIdx.x;
  size_t total = (size_t)N * HF;
  if (idx >= total) return;
  int hf = (int)(idx % HF);
  buf[idx] = fmaxf(buf[idx] + bias[hf], 0.0f);
}

// ---------------- global max pool ----------------
__global__ void k_pool_max(const float* __restrict__ h, const int* __restrict__ batch,
                           int* __restrict__ enc, int N, int F) {
  size_t idx = (size_t)blockIdx.x * blockDim.x + threadIdx.x;
  size_t total = (size_t)N * F;
  if (idx >= total) return;
  int f = (int)(idx % F);
  size_t n = idx / F;
  int g = batch[n];
  atomicMax(&enc[(size_t)g * F + f], enc_f32(h[idx]));
}
__global__ void k_pool_decode(const int* __restrict__ enc, float* __restrict__ out, size_t n) {
  size_t i = (size_t)blockIdx.x * blockDim.x + threadIdx.x;
  if (i >= n) return;
  float v = dec_f32(enc[i]);
  out[i] = __builtin_isfinite(v) ? v : 0.0f;
}

// ---------------- host orchestration ----------------
static inline unsigned nblk(size_t total, int bs) {
  return (unsigned)((total + (size_t)bs - 1) / (size_t)bs);
}

extern "C" void kernel_launch(void* const* d_in, const int* in_sizes, int n_in,
                              void* d_out, int out_size, void* d_ws, size_t ws_size,
                              hipStream_t stream) {
  (void)in_sizes; (void)n_in; (void)out_size; (void)ws_size;
  const float* x    = (const float*)d_in[0];
  const int*   ei   = (const int*)d_in[1];
  const int*   batch= (const int*)d_in[2];
  const float* W1   = (const float*)d_in[4];
  const float* as1  = (const float*)d_in[5];
  const float* ad1  = (const float*)d_in[6];
  const float* b1   = (const float*)d_in[7];
  const float* W2   = (const float*)d_in[8];
  const float* as2  = (const float*)d_in[9];
  const float* ad2  = (const float*)d_in[10];
  const float* b2   = (const float*)d_in[11];
  const float* fcW  = (const float*)d_in[12];
  const float* fcb  = (const float*)d_in[13];
  float* out        = (float*)d_out;

  // ---- workspace bump layout (256B aligned) ----
  char* base = (char*)d_ws;
  size_t off = 0;
  auto alloc = [&](size_t bytes) -> char* {
    off = (off + 255) & ~(size_t)255;
    char* p = base + off;
    off += bytes;
    return p;
  };
  float*          h1      = (float*)alloc((size_t)NN * HF1 * 4);     // x@W1
  float*          agg1    = (float*)alloc((size_t)NN * HF1 * 4);     // layer1 agg -> x2
  unsigned short* abf     = (unsigned short*)alloc((size_t)NN * KP2 * 2); // bf16 A (both layers)
  unsigned short* btbf    = (unsigned short*)alloc((size_t)256 * 1024);   // bf16 Bt (all weights)
  float*          asrc    = (float*)alloc((size_t)NN * H1 * 4);
  float*          adst    = (float*)alloc((size_t)NN * H1 * 4);
  int*            maxb    = (int*)alloc((size_t)NN * H1 * 4);
  float*          denom   = (float*)alloc((size_t)NN * H1 * 4);
  float*          ebuf    = (float*)alloc((size_t)ET * H1 * 4);
  float*          h2      = (float*)alloc((size_t)NN * OUT2 * 4);
  float*          agg2    = (float*)alloc((size_t)NN * OUT2 * 4);
  int*            penc    = (int*)alloc((size_t)GG * OUT2 * 4);
  float*          pooled  = (float*)alloc((size_t)GG * OUT2 * 4);
  unsigned short* pbf     = (unsigned short*)alloc((size_t)GG * KPF * 2);

  const int BS = 256;

  // ================= layer 1: GATConv(78 -> 78, heads=10) =================
  // h1 = x @ W1   (bf16 WMMA, LDS-staged weight panel)
  k_conv_a <<<nblk((size_t)NN * KP1, BS), BS, 0, stream>>>(x, abf, NN, FIN, KP1);
  k_conv_bt<<<nblk((size_t)HF1 * KP1, BS), BS, 0, stream>>>(W1, btbf, FIN, HF1, KP1);
  {
    dim3 g((NN + 127) / 128, (HF1 + 63) / 64);
    k_wmma_gemm_bf16<<<g, BS, 64 * KP1 * 2, stream>>>(abf, btbf, h1, nullptr,
                                                      NN, HF1, KP1, 0);
  }
  k_alpha_proj<<<nblk((size_t)NN * H1, BS), BS, 0, stream>>>(h1, as1, asrc, NN, H1, FIN);
  k_alpha_proj<<<nblk((size_t)NN * H1, BS), BS, 0, stream>>>(h1, ad1, adst, NN, H1, FIN);
  k_fill_i32<<<nblk((size_t)NN * H1, BS), BS, 0, stream>>>(maxb, ENC_NEG_INF, (size_t)NN * H1);
  k_fill_f32<<<nblk((size_t)NN * H1, BS), BS, 0, stream>>>(denom, 0.0f, (size_t)NN * H1);
  k_fill_f32<<<nblk((size_t)NN * HF1, BS), BS, 0, stream>>>(agg1, 0.0f, (size_t)NN * HF1);
  k_edge_logits<<<nblk((size_t)ET * H1, BS), BS, 0, stream>>>(ei, asrc, adst, ebuf, maxb, H1);
  k_edge_exp   <<<nblk((size_t)ET * H1, BS), BS, 0, stream>>>(ei, ebuf, maxb, denom, H1);
  k_edge_scatter<<<16384, BS, 0, stream>>>(ei, ebuf, denom, h1, agg1, H1, FIN);
  k_bias_relu<<<nblk((size_t)NN * HF1, BS), BS, 0, stream>>>(agg1, b1, NN, HF1);

  // ================= layer 2: GATConv(780 -> 128, heads=1) =================
  k_conv_a <<<nblk((size_t)NN * KP2, BS), BS, 0, stream>>>(agg1, abf, NN, HF1, KP2);
  k_conv_bt<<<nblk((size_t)OUT2 * KP2, BS), BS, 0, stream>>>(W2, btbf, HF1, OUT2, KP2);
  {
    dim3 g((NN + 127) / 128, (OUT2 + 63) / 64);
    k_wmma_gemm_bf16<<<g, BS, 64 * KP2 * 2, stream>>>(abf, btbf, h2, nullptr,
                                                      NN, OUT2, KP2, 0);
  }
  k_alpha_proj<<<nblk((size_t)NN, BS), BS, 0, stream>>>(h2, as2, asrc, NN, 1, OUT2);
  k_alpha_proj<<<nblk((size_t)NN, BS), BS, 0, stream>>>(h2, ad2, adst, NN, 1, OUT2);
  k_fill_i32<<<nblk((size_t)NN, BS), BS, 0, stream>>>(maxb, ENC_NEG_INF, (size_t)NN);
  k_fill_f32<<<nblk((size_t)NN, BS), BS, 0, stream>>>(denom, 0.0f, (size_t)NN);
  k_fill_f32<<<nblk((size_t)NN * OUT2, BS), BS, 0, stream>>>(agg2, 0.0f, (size_t)NN * OUT2);
  k_edge_logits<<<nblk((size_t)ET, BS), BS, 0, stream>>>(ei, asrc, adst, ebuf, maxb, 1);
  k_edge_exp   <<<nblk((size_t)ET, BS), BS, 0, stream>>>(ei, ebuf, maxb, denom, 1);
  k_edge_scatter<<<16384, BS, 0, stream>>>(ei, ebuf, denom, h2, agg2, 1, OUT2);
  k_bias_relu<<<nblk((size_t)NN * OUT2, BS), BS, 0, stream>>>(agg2, b2, NN, OUT2);

  // ================= global max pool + FC =================
  k_fill_i32<<<nblk((size_t)GG * OUT2, BS), BS, 0, stream>>>(penc, ENC_NEG_INF, (size_t)GG * OUT2);
  k_pool_max<<<nblk((size_t)NN * OUT2, BS), BS, 0, stream>>>(agg2, batch, penc, NN, OUT2);
  k_pool_decode<<<nblk((size_t)GG * OUT2, BS), BS, 0, stream>>>(penc, pooled, (size_t)GG * OUT2);

  k_conv_a <<<nblk((size_t)GG * KPF, BS), BS, 0, stream>>>(pooled, pbf, GG, OUT2, KPF);
  k_conv_bt<<<nblk((size_t)OUT2 * KPF, BS), BS, 0, stream>>>(fcW, btbf, OUT2, OUT2, KPF);
  {
    dim3 g((GG + 127) / 128, (OUT2 + 63) / 64);
    k_wmma_gemm_bf16<<<g, BS, 64 * KPF * 2, stream>>>(pbf, btbf, out, fcb,
                                                      GG, OUT2, KPF, 1);
  }
}
